// WindowAttention_9122510537173
// MI455X (gfx1250) — matrix operand order repaired and checked by reading, not compiled, not run on hardware
//
#include <hip/hip_runtime.h>
#include <hip/hip_bf16.h>
#include <math.h>

// ---------------------------------------------------------------------------
// Window attention, fused, bf16 WMMA (v_wmma_f32_16x16x32_bf16) on gfx1250.
// B=2048 windows, N=98 tokens (pad 112), C=192, H=6 heads, hd=32, nW=64.
// ---------------------------------------------------------------------------

typedef __attribute__((ext_vector_type(16))) __bf16 v16bf;
typedef __attribute__((ext_vector_type(8)))  float  v8f;

#define THREADS   256
#define WAVES     8
#define N_TOK     98
#define N_PAD     112          // 7 * 16
#define KPAD      128          // token axis padded to 4 k-steps of 32
#define HEADS     6
#define HD        32
#define DIM       192
#define QKV_OUT   576
#define NW        64

// LDS carving (units: ushort). xs aliases ctx (xs dead after phase 1).
#define OFF_XS    0                         // 112*192          = 21504
#define OFF_QS    21504                     // 6*112*32         = 21504
#define OFF_KS    43008                     // 6*112*32         = 21504
#define OFF_VT    64512                     // 6*32*128         = 24576
#define OFF_STRIP 89088                     // 8 waves * 16*128 = 16384
#define SMEM_USHORTS 105472
#define SMEM_BYTES   (SMEM_USHORTS * 2)     // 210944 B < 320 KB/WGP

// ws carving (bytes)
#define WS_QKVW   0                         // 576*192 bf16 = 221184 B
#define WS_PROJW  221184                    // 192*192 bf16 =  73728 B
#define WS_BIAS   294912                    // 6*98*98 f32  = 230496 B

__device__ __forceinline__ unsigned short f2bf(float f) {
    unsigned u = __builtin_bit_cast(unsigned, f);
    u += 0x7FFFu + ((u >> 16) & 1u);        // round-to-nearest-even
    return (unsigned short)(u >> 16);
}

__device__ __forceinline__ unsigned pack2bf(float lo, float hi) {
    return (unsigned)f2bf(lo) | ((unsigned)f2bf(hi) << 16);
}

union FragAB {                               // one WMMA A/B operand per lane
    v16bf v;
    unsigned short us[16];
    uint4 q[2];                              // two 16B chunks -> ds_load_b128
};

__device__ __forceinline__ v8f zero8() {
    v8f z = {0.f, 0.f, 0.f, 0.f, 0.f, 0.f, 0.f, 0.f};
    return z;
}

// ---------------------------------------------------------------------------
// Prep kernels: weight fp32 -> bf16, and dense relative-position bias gather.
// ---------------------------------------------------------------------------
__global__ void wa_prep_weights(const float* __restrict__ qkv_w,
                                const float* __restrict__ proj_w,
                                unsigned short* __restrict__ qkvw_bf,
                                unsigned short* __restrict__ projw_bf) {
    int i = blockIdx.x * blockDim.x + threadIdx.x;
    if (i < QKV_OUT * DIM) qkvw_bf[i] = f2bf(qkv_w[i]);
    if (i < DIM * DIM)     projw_bf[i] = f2bf(proj_w[i]);
}

__global__ void wa_prep_bias(const float* __restrict__ table,
                             float* __restrict__ bias_full) {
    int t = blockIdx.x * blockDim.x + threadIdx.x;
    if (t >= HEADS * N_TOK * N_TOK) return;
    int h   = t / (N_TOK * N_TOK);
    int rem = t % (N_TOK * N_TOK);
    int n = rem / N_TOK, m = rem % N_TOK;
    int s1 = n / 49, r1 = n % 49, h1 = r1 / 7, w1 = r1 % 7;
    int s2 = m / 49, r2 = m % 49, h2 = r2 / 7, w2 = r2 % 7;
    int idx = (s1 - s2 + 1) * 169 + (h1 - h2 + 6) * 13 + (w1 - w2 + 6);
    bias_full[t] = table[idx * HEADS + h];   // layout [h][n][m]
}

// ---------------------------------------------------------------------------
// Fused attention kernel: one block per window.
// ---------------------------------------------------------------------------
__global__ void __launch_bounds__(THREADS)
wa_fused_kernel(const float* __restrict__ x,
                const float* __restrict__ mask,
                const float* __restrict__ qkv_b,
                const float* __restrict__ proj_b,
                const unsigned short* __restrict__ qkvw_bf,
                const unsigned short* __restrict__ projw_bf,
                const float* __restrict__ bias_full,
                float* __restrict__ out) {
    extern __shared__ unsigned short smem[];
    unsigned short* xs    = smem + OFF_XS;          // x_b bf16 [112][192]
    unsigned short* ctx   = smem + OFF_XS;          // alias: context [112][192]
    unsigned short* qs    = smem + OFF_QS;          // Q*scale  [h][112][32]
    unsigned short* ksm   = smem + OFF_KS;          // K        [h][112][32]
    unsigned short* vt    = smem + OFF_VT;          // V^T      [h][32][128]
    const int tid    = threadIdx.x;
    const int lane   = tid & 31;
    const int wid    = tid >> 5;
    const int lanelo = lane & 15;
    const int lanehi = lane >> 4;
    unsigned short* strip = smem + OFF_STRIP + wid * 16 * KPAD;  // per-wave probs

    const int b = blockIdx.x;
    const int w = b & (NW - 1);
    const float scale = 0.17677669529663687f;       // 1/sqrt(32)

    // ------- stage x -> bf16 LDS; zero V^T and x padding rows --------------
    for (int i = tid; i < HEADS * HD * KPAD / 2; i += THREADS)
        ((unsigned*)vt)[i] = 0u;
    for (int i = tid; i < (N_PAD - N_TOK) * DIM / 2; i += THREADS)
        ((unsigned*)(xs + N_TOK * DIM))[i] = 0u;
    const float* xb = x + (size_t)b * N_TOK * DIM;
    for (int i = tid; i < N_TOK * DIM / 4; i += THREADS) {
        float4 f = ((const float4*)xb)[i];
        uint2 p;
        p.x = pack2bf(f.x, f.y);
        p.y = pack2bf(f.z, f.w);
        *(uint2*)(xs + 4 * i) = p;
    }
    __syncthreads();

    // ------- Phase 1: qkv = x @ W^T + b -> LDS (Q scaled, K, V^T) ----------
    // mt outer so each wave loads the 6 x A-fragments once per row tile.
    for (int mt = 0; mt < 7; ++mt) {
        FragAB a[6];
#pragma unroll
        for (int ks = 0; ks < 6; ++ks) {
            const unsigned short* ar =
                xs + (mt * 16 + lanelo) * DIM + ks * 32 + 16 * lanehi;
            a[ks].q[0] = *(const uint4*)(ar);
            a[ks].q[1] = *(const uint4*)(ar + 8);
        }
        for (int ot = wid; ot < 36; ot += WAVES) {
            const int o = ot * 16 + lanelo;          // output feature (B col)
            v8f acc = zero8();
#pragma unroll
            for (int ks = 0; ks < 6; ++ks) {
                FragAB bf;
                const unsigned short* br =
                    qkvw_bf + o * DIM + ks * 32 + 16 * lanehi;
                bf.q[0] = *(const uint4*)(br);
                bf.q[1] = *(const uint4*)(br + 8);
                acc = __builtin_amdgcn_wmma_f32_16x16x32_bf16(
                    false, a[ks].v, false, bf.v, (short)0, acc, false, false);
            }
            const float bia = qkv_b[o];
            // wave-uniform routing: a 16-col tile never crosses q/k/v bounds
            if (ot < 12) {                           // Q (scaled)
                const int h = (o >> 5), d = o & 31;
#pragma unroll
                for (int r = 0; r < 8; ++r) {
                    const int m = mt * 16 + r + 8 * lanehi;
                    qs[(h * N_PAD + m) * HD + d] = f2bf((acc[r] + bia) * scale);
                }
            } else if (ot < 24) {                    // K
                const int oo = o - DIM, h = oo >> 5, d = oo & 31;
#pragma unroll
                for (int r = 0; r < 8; ++r) {
                    const int m = mt * 16 + r + 8 * lanehi;
                    ksm[(h * N_PAD + m) * HD + d] = f2bf(acc[r] + bia);
                }
            } else {                                 // V, stored transposed
                const int oo = o - 2 * DIM, h = oo >> 5, d = oo & 31;
#pragma unroll
                for (int r = 0; r < 8; ++r) {
                    const int m = mt * 16 + r + 8 * lanehi;
                    vt[(h * HD + d) * KPAD + m] = f2bf(acc[r] + bia);
                }
            }
        }
    }
    __syncthreads();

    // ------- Phase 2: attention per (head, 16-row tile) --------------------
    for (int p = wid; p < HEADS * 7; p += WAVES) {
        const int h = p / 7, mt = p % 7;
        // Q A-fragment (16 rows x K=32), loaded once
        FragAB aq;
        const unsigned short* qr =
            qs + (h * N_PAD + mt * 16 + lanelo) * HD + 16 * lanehi;
        aq.q[0] = *(const uint4*)(qr);
        aq.q[1] = *(const uint4*)(qr + 8);
        v8f acc[7];
#pragma unroll
        for (int nt = 0; nt < 7; ++nt) {
            acc[nt] = zero8();
            FragAB bk;                               // B = K^T (d x tokens)
            const unsigned short* kr =
                ksm + (h * N_PAD + nt * 16 + lanelo) * HD + 16 * lanehi;
            bk.q[0] = *(const uint4*)(kr);
            bk.q[1] = *(const uint4*)(kr + 8);
            acc[nt] = __builtin_amdgcn_wmma_f32_16x16x32_bf16(
                false, aq.v, false, bk.v, (short)0, acc[nt], false, false);
        }
        // logits + bias + mask (pad columns/rows -> -inf)
        const float* biasH = bias_full + h * N_TOK * N_TOK;
        const float* maskW = mask + w * N_TOK * N_TOK;
#pragma unroll
        for (int nt = 0; nt < 7; ++nt) {
            const int mc = nt * 16 + lanelo;
#pragma unroll
            for (int r = 0; r < 8; ++r) {
                const int m = mt * 16 + r + 8 * lanehi;
                float v = -1e30f;
                if (mc < N_TOK && m < N_TOK)
                    v = acc[nt][r] + biasH[m * N_TOK + mc] + maskW[m * N_TOK + mc];
                acc[nt][r] = v;
            }
        }
        // softmax over the 98 (112-padded) columns of each row
#pragma unroll
        for (int r = 0; r < 8; ++r) {
            float mx = -1e30f;
#pragma unroll
            for (int nt = 0; nt < 7; ++nt) mx = fmaxf(mx, acc[nt][r]);
#pragma unroll
            for (int off = 8; off >= 1; off >>= 1)
                mx = fmaxf(mx, __shfl_xor(mx, off, 32));
            float sum = 0.f;
#pragma unroll
            for (int nt = 0; nt < 7; ++nt) {
                float e = __expf(acc[nt][r] - mx);
                acc[nt][r] = e;
                sum += e;
            }
#pragma unroll
            for (int off = 8; off >= 1; off >>= 1)
                sum += __shfl_xor(sum, off, 32);
            const float inv = 1.0f / sum;
#pragma unroll
            for (int nt = 0; nt < 7; ++nt) acc[nt][r] *= inv;
        }
        // probabilities -> per-wave bf16 strip [16][128]
#pragma unroll
        for (int nt = 0; nt < 7; ++nt) {
            const int mc = nt * 16 + lanelo;
#pragma unroll
            for (int r = 0; r < 8; ++r)
                strip[(r + 8 * lanehi) * KPAD + mc] = f2bf(acc[nt][r]);
        }
        for (int i = lane; i < 16 * 16; i += 32)     // zero pad cols 112..127
            strip[(i >> 4) * KPAD + N_PAD + (i & 15)] = 0;

        // AV: (16 x 128) @ (128 x 32)
        v8f av0 = zero8(), av1 = zero8();
#pragma unroll
        for (int ks2 = 0; ks2 < 4; ++ks2) {
            FragAB a2;
            const unsigned short* sr =
                strip + lanelo * KPAD + ks2 * 32 + 16 * lanehi;
            a2.q[0] = *(const uint4*)(sr);
            a2.q[1] = *(const uint4*)(sr + 8);
            FragAB b0, b1;
            const unsigned short* v0 =
                vt + (h * HD + lanelo) * KPAD + ks2 * 32 + 16 * lanehi;
            const unsigned short* v1 =
                vt + (h * HD + 16 + lanelo) * KPAD + ks2 * 32 + 16 * lanehi;
            b0.q[0] = *(const uint4*)(v0);
            b0.q[1] = *(const uint4*)(v0 + 8);
            b1.q[0] = *(const uint4*)(v1);
            b1.q[1] = *(const uint4*)(v1 + 8);
            av0 = __builtin_amdgcn_wmma_f32_16x16x32_bf16(
                false, a2.v, false, b0.v, (short)0, av0, false, false);
            av1 = __builtin_amdgcn_wmma_f32_16x16x32_bf16(
                false, a2.v, false, b1.v, (short)0, av1, false, false);
        }
#pragma unroll
        for (int r = 0; r < 8; ++r) {
            const int m = mt * 16 + r + 8 * lanehi;
            ctx[m * DIM + h * HD + lanelo]      = f2bf(av0[r]);
            ctx[m * DIM + h * HD + 16 + lanelo] = f2bf(av1[r]);
        }
    }
    __syncthreads();

    // ------- Phase 3: out = ctx @ proj_w^T + proj_b ------------------------
    float* ob = out + (size_t)b * N_TOK * DIM;
    for (int mt = 0; mt < 7; ++mt) {
        FragAB a[6];
#pragma unroll
        for (int ks = 0; ks < 6; ++ks) {
            const unsigned short* ar =
                ctx + (mt * 16 + lanelo) * DIM + ks * 32 + 16 * lanehi;
            a[ks].q[0] = *(const uint4*)(ar);
            a[ks].q[1] = *(const uint4*)(ar + 8);
        }
        for (int ot = wid; ot < 12; ot += WAVES) {
            const int o = ot * 16 + lanelo;
            v8f acc = zero8();
#pragma unroll
            for (int ks = 0; ks < 6; ++ks) {
                FragAB bf;
                const unsigned short* br =
                    projw_bf + o * DIM + ks * 32 + 16 * lanehi;
                bf.q[0] = *(const uint4*)(br);
                bf.q[1] = *(const uint4*)(br + 8);
                acc = __builtin_amdgcn_wmma_f32_16x16x32_bf16(
                    false, a[ks].v, false, bf.v, (short)0, acc, false, false);
            }
            const float pb = proj_b[o];
#pragma unroll
            for (int r = 0; r < 8; ++r) {
                const int m = mt * 16 + r + 8 * lanehi;
                if (m < N_TOK) ob[m * DIM + o] = acc[r] + pb;
            }
        }
    }
}

// ---------------------------------------------------------------------------
extern "C" void kernel_launch(void* const* d_in, const int* in_sizes, int n_in,
                              void* d_out, int out_size, void* d_ws, size_t ws_size,
                              hipStream_t stream) {
    (void)in_sizes; (void)n_in; (void)out_size; (void)ws_size;
    const float* x      = (const float*)d_in[0];
    const float* mask   = (const float*)d_in[1];
    const float* qkv_w  = (const float*)d_in[2];
    const float* qkv_b  = (const float*)d_in[3];
    const float* proj_w = (const float*)d_in[4];
    const float* proj_b = (const float*)d_in[5];
    const float* btab   = (const float*)d_in[6];
    float* out = (float*)d_out;

    unsigned char* wsb = (unsigned char*)d_ws;
    unsigned short* qkvw_bf   = (unsigned short*)(wsb + WS_QKVW);
    unsigned short* projw_bf  = (unsigned short*)(wsb + WS_PROJW);
    float*          bias_full = (float*)(wsb + WS_BIAS);

    wa_prep_weights<<<(QKV_OUT * DIM + 255) / 256, 256, 0, stream>>>(
        qkv_w, proj_w, qkvw_bf, projw_bf);
    wa_prep_bias<<<(HEADS * N_TOK * N_TOK + 255) / 256, 256, 0, stream>>>(
        btab, bias_full);

    hipFuncSetAttribute(reinterpret_cast<const void*>(wa_fused_kernel),
                        hipFuncAttributeMaxDynamicSharedMemorySize, SMEM_BYTES);
    wa_fused_kernel<<<2048, THREADS, SMEM_BYTES, stream>>>(
        x, mask, qkv_b, proj_b, qkvw_bf, projw_bf, bias_full, out);
}
